// BitConv1d_68118181314691
// MI455X (gfx1250) — 8-bit matrix operand order repaired and checked by reading, compiled, not run on hardware
//
#include <hip/hip_runtime.h>
#include <hip/hip_bf16.h>
#include <math.h>

#define IN_CH   512
#define OUT_CH  512
#define T_LEN   8192
#define BATCH   8
#define KSIZE   7
#define T_PAD   (T_LEN + 16)

typedef int v8i __attribute__((ext_vector_type(8)));

#define WS_AMAX_OFF   0
#define WS_WSCALE_OFF 4
#define WS_WPART_OFF  64
#define WS_RMS_OFF    4096
#define WS_W8_OFF     (4096 + 65536 * 4)
#define WS_XQ_OFF     (WS_W8_OFF + 1835008)

#define LDS_PITCH 80
#define LDS_BUF   (520 * LDS_PITCH)

__device__ __forceinline__ float sanitize_f(float v) {
    if (isnan(v)) return 0.f;
    if (isinf(v)) return v > 0.f ? 1e4f : -1e4f;
    return v;
}

__device__ __forceinline__ void async_load_b128(unsigned lds_off, const void* gaddr) {
    asm volatile("global_load_async_to_lds_b128 %0, %1, off"
                 :: "v"(lds_off), "v"((unsigned long long)(size_t)gaddr)
                 : "memory");
}
__device__ __forceinline__ void wait_asynccnt0() {
    asm volatile("s_wait_asynccnt 0x0" ::: "memory");
}

__global__ void k_rms_amax(const float* __restrict__ x, const float* __restrict__ gamma,
                           float* __restrict__ rms_out, unsigned int* __restrict__ amax_bits) {
    int idx = blockIdx.x * 256 + threadIdx.x;
    int b = idx >> 13, t = idx & (T_LEN - 1);
    const float* xp = x + (size_t)b * IN_CH * T_LEN + t;
    float ss = 0.f, ma = 0.f;
    for (int c = 0; c < IN_CH; ++c) {
        float v = xp[(size_t)c * T_LEN];
        ss = fmaf(v, v, ss);
        ma = fmaxf(ma, fabsf(v * gamma[c]));
    }
    float rms = rsqrtf(ss * (1.f / IN_CH) + 1e-6f);
    rms_out[idx] = rms;
    float am = ma * rms;
    if (isnan(am)) am = 0.f;
    if (isinf(am)) am = 1e4f;
    __shared__ float sm[256];
    sm[threadIdx.x] = am;
    __syncthreads();
    for (int s = 128; s > 0; s >>= 1) {
        if (threadIdx.x < s) sm[threadIdx.x] = fmaxf(sm[threadIdx.x], sm[threadIdx.x + s]);
        __syncthreads();
    }
    if (threadIdx.x == 0) atomicMax(amax_bits, __float_as_uint(sm[0]));
}

__global__ void k_wabs_partial(const float* __restrict__ w, float* __restrict__ part) {
    const int N = OUT_CH * IN_CH * KSIZE;
    float s = 0.f;
    for (int i = blockIdx.x * 256 + threadIdx.x; i < N; i += 256 * 256) s += fabsf(w[i]);
    __shared__ float sm[256];
    sm[threadIdx.x] = s;
    __syncthreads();
    for (int st = 128; st > 0; st >>= 1) {
        if (threadIdx.x < st) sm[threadIdx.x] += sm[threadIdx.x + st];
        __syncthreads();
    }
    if (threadIdx.x == 0) part[blockIdx.x] = sm[0];
}

__global__ void k_wscale(const float* __restrict__ part, float* __restrict__ wscale) {
    __shared__ float sm[256];
    sm[threadIdx.x] = part[threadIdx.x];
    __syncthreads();
    for (int st = 128; st > 0; st >>= 1) {
        if (threadIdx.x < st) sm[threadIdx.x] += sm[threadIdx.x + st];
        __syncthreads();
    }
    if (threadIdx.x == 0)
        wscale[0] = fmaxf(sm[0] / (float)(OUT_CH * IN_CH * KSIZE), 1e-4f);
}

__global__ void k_wpack(const float* __restrict__ w, const float* __restrict__ wscale,
                        unsigned int* __restrict__ w8) {
    int gid = blockIdx.x * 256 + threadIdx.x;
    float inv = 1.f / wscale[0];
    int v    = gid & 7;
    int lane = (gid >> 3) & 31;
    int frag = gid >> 8;
    int icC = frag & 7, ocT = (frag >> 3) & 31, k = frag >> 8;
    int m = lane & 15, hi = lane >> 4;
    int kbyte = ((v >> 2) << 5) + (hi << 4) + ((v & 3) << 2);
    int ic = icC * 64 + kbyte;
    int oc = ocT * 16 + m;
    unsigned int d = 0;
    #pragma unroll
    for (int j = 0; j < 4; ++j) {
        float wn = w[((size_t)oc * IN_CH + ic + j) * KSIZE + k] * inv;
        int q = (int)rintf(fminf(fmaxf(wn, -1.f), 1.f));
        d |= ((unsigned int)(q & 0xFF)) << (8 * j);
    }
    w8[gid] = d;
}

__global__ void k_actq(const float* __restrict__ x, const float* __restrict__ gamma,
                       const float* __restrict__ rms, const unsigned int* __restrict__ amax_bits,
                       char* __restrict__ xq) {
    __shared__ char sq[64 * 512];
    int wg = blockIdx.x;
    int b = wg >> 7, t0 = (wg & 127) * 64;
    int tid = threadIdx.x;
    int cc = tid >> 6, tt = tid & 63;
    float scale = fmaxf(__uint_as_float(amax_bits[0]), 1e-5f);
    float qmul = 127.0f / scale;
    float r = rms[b * T_LEN + t0 + tt];
    const float* xb = x + (size_t)b * IN_CH * T_LEN + t0 + tt;
    for (int c0 = 0; c0 < IN_CH; c0 += 4) {
        int c = c0 + cc;
        float vv = xb[(size_t)c * T_LEN];
        float xn = sanitize_f(vv * r * gamma[c]);
        float xs = xn * qmul;
        int q = (int)rintf(fminf(fmaxf(xs, -128.f), 127.f));
        sq[tt * 512 + c] = (char)q;
    }
    __syncthreads();
    int row = tid >> 2, seg = tid & 3;
    const int4* src = (const int4*)(sq + row * 512 + seg * 128);
    int4* dst = (int4*)(xq + ((size_t)b * T_PAD + 8 + t0 + row) * 512 + seg * 128);
    #pragma unroll
    for (int i = 0; i < 8; ++i) dst[i] = src[i];
}

__global__ void k_conv(const char* __restrict__ xq, const unsigned int* __restrict__ w8,
                       const unsigned int* __restrict__ amax_bits,
                       const float* __restrict__ wscale, float* __restrict__ out) {
    __shared__ char xt[2][LDS_BUF];
    const int t0  = blockIdx.x * 512;
    const int ocT = blockIdx.y;
    const int b   = blockIdx.z;
    const int tid = threadIdx.x;
    const int wave = tid >> 5, lane = tid & 31;
    const int col = lane & 15, hi = lane >> 4;

    const unsigned lds_base =
        (unsigned)(size_t)(__attribute__((address_space(3))) char*)&xt[0][0];

    v8i acc[4] = {};

    const char* xbase = xq + ((size_t)b * T_PAD + (t0 + 5)) * 512;

    auto stage_async = [&](int chunk, int buf) {
        unsigned lbase = lds_base + (unsigned)buf * LDS_BUF + (tid & 3) * 16;
        const char* gbase = xbase + chunk * 64 + (tid & 3) * 16;
        for (int r = tid >> 2; r < 518; r += 64)
            async_load_b128(lbase + (unsigned)r * LDS_PITCH, gbase + (size_t)r * 512);
    };

    stage_async(0, 0);
    wait_asynccnt0();
    __syncthreads();

    for (int chunk = 0; chunk < 8; ++chunk) {
        const int cur = chunk & 1;
        if (chunk < 7) stage_async(chunk + 1, cur ^ 1);

        const char* xcur = &xt[cur][0];
        #pragma unroll
        for (int k = 0; k < KSIZE; ++k) {
            const int4* wp = (const int4*)((const char*)w8 +
                (((size_t)((k * 32 + ocT) * 8 + chunk)) << 10) + lane * 32);
            int4 a0 = wp[0], a1 = wp[1];
            v8i A;
            A[0] = a0.x; A[1] = a0.y; A[2] = a0.z; A[3] = a0.w;
            A[4] = a1.x; A[5] = a1.y; A[6] = a1.z; A[7] = a1.w;
            #pragma unroll
            for (int j = 0; j < 4; ++j) {
                int rloc = wave * 64 + j * 16 + col + k;
                const char* bb = xcur + rloc * LDS_PITCH + hi * 16;
                int4 b0 = *(const int4*)(bb);
                int4 b1 = *(const int4*)(bb + 32);
                v8i B;
                B[0] = b0.x; B[1] = b0.y; B[2] = b0.z; B[3] = b0.w;
                B[4] = b1.x; B[5] = b1.y; B[6] = b1.z; B[7] = b1.w;
                acc[j] = __builtin_amdgcn_wmma_i32_16x16x64_iu8(
                    true, A, true, B, acc[j], false, false);
            }
        }
        wait_asynccnt0();
        __syncthreads();
    }

    float mul = fmaxf(__uint_as_float(amax_bits[0]), 1e-5f) * (1.0f / 127.0f) * wscale[0];
    #pragma unroll
    for (int j = 0; j < 4; ++j) {
        int t = t0 + wave * 64 + j * 16 + col;
        #pragma unroll
        for (int rr = 0; rr < 8; ++rr) {
            int oc = ocT * 16 + rr + hi * 8;
            out[((size_t)b * OUT_CH + oc) * T_LEN + t] = (float)acc[j][rr] * mul;
        }
    }
}

extern "C" void kernel_launch(void* const* d_in, const int* in_sizes, int n_in,
                              void* d_out, int out_size, void* d_ws, size_t ws_size,
                              hipStream_t stream) {
    const float* x     = (const float*)d_in[0];
    const float* w     = (const float*)d_in[1];
    const float* gamma = (const float*)d_in[2];
    float* out = (float*)d_out;
    char* ws = (char*)d_ws;

    unsigned int* amax = (unsigned int*)(ws + WS_AMAX_OFF);
    float* wscale      = (float*)(ws + WS_WSCALE_OFF);
    float* wpart       = (float*)(ws + WS_WPART_OFF);
    float* rms         = (float*)(ws + WS_RMS_OFF);
    unsigned int* w8   = (unsigned int*)(ws + WS_W8_OFF);
    char* xq           = ws + WS_XQ_OFF;

    hipMemsetAsync(ws, 0, 4096, stream);
    hipMemsetAsync(xq, 0, (size_t)BATCH * T_PAD * 512, stream);

    k_rms_amax<<<256, 256, 0, stream>>>(x, gamma, rms, amax);
    k_wabs_partial<<<256, 256, 0, stream>>>(w, wpart);
    k_wscale<<<1, 256, 0, stream>>>(wpart, wscale);
    k_wpack<<<1792, 256, 0, stream>>>(w, wscale, w8);
    k_actq<<<1024, 256, 0, stream>>>(x, gamma, rms, amax, xq);

    dim3 grid(16, 32, 8);
    k_conv<<<grid, 256, 0, stream>>>(xq, w8, amax, wscale, out);
}
